// MultiheadAttention_42923903156951
// MI455X (gfx1250) — matrix operand order repaired and checked by reading, not compiled, not run on hardware
//
#include <hip/hip_runtime.h>
#include <hip/hip_bf16.h>

// ---------- types matching CDNA5 WMMA builtin signatures ----------
typedef __bf16 bf16_t;
typedef __attribute__((ext_vector_type(16))) __bf16 v16bf;
typedef __attribute__((ext_vector_type(8)))  __bf16 v8bf;
typedef __attribute__((ext_vector_type(8)))  float  v8f;

#define NB     16
#define NHEADS 6
#define NPOS   1024
#define CDIM   384
#define HDIM   64

__device__ __forceinline__ v8f wmma_bf16(v16bf a, v16bf b, v8f c) {
  // D = A(16x32) * B(32x16) + C(16x16 f32)
  return __builtin_amdgcn_wmma_f32_16x16x32_bf16(false, a, false, b, (short)0, c, false, false);
}

// A fragment: 16x32 (MxK). Lane group g, lane r: rows M=r,
// elements 0..7  -> K = k0 + 16g + 0..7   (VGPR0..3)
// elements 8..15 -> K = k0 + 16g + 8..15 (VGPR4..7)
__device__ __forceinline__ v16bf load_afrag(const bf16_t* __restrict__ base, int ld, int row0, int k0) {
  const int lane = threadIdx.x & 31;
  const int g = lane >> 4, r = lane & 15;
  const bf16_t* p = base + (size_t)(row0 + r) * ld + (k0 + 16 * g);
  v8bf lo = *(const v8bf*)(p);
  v8bf hi = *(const v8bf*)(p + 8);
  v16bf out;
#pragma unroll
  for (int i = 0; i < 8; ++i) { out[i] = lo[i]; out[i + 8] = hi[i]; }
  return out;
}

// B fragment: 32x16 (KxN), column n stored K-contiguous (row n of an NT weight).
// Lane group g, lane r: column N=r, element e -> K = k0 + 16g + e.
__device__ __forceinline__ v16bf load_bfrag(const bf16_t* __restrict__ base, int ld, int n0, int k0) {
  const int lane = threadIdx.x & 31;
  const int g = lane >> 4, r = lane & 15;
  return *(const v16bf*)(base + (size_t)(n0 + r) * ld + (k0 + 16 * g));
}

// ---------- kernel 0: fp32 -> bf16 weight conversion (Wq/Wk/Wv/Wp) ----------
__global__ void cvt_weights(const float* __restrict__ a, const float* __restrict__ b,
                            const float* __restrict__ c, const float* __restrict__ d,
                            bf16_t* __restrict__ oa, bf16_t* __restrict__ ob,
                            bf16_t* __restrict__ oc, bf16_t* __restrict__ od) {
  int i = blockIdx.x * blockDim.x + threadIdx.x;  // grid exactly 384*384
  oa[i] = (bf16_t)a[i]; ob[i] = (bf16_t)b[i]; oc[i] = (bf16_t)c[i]; od[i] = (bf16_t)d[i];
}

// ---------- kernel 1: fused depthwise 3x3 conv + BN for q/k/v, fp32 -> bf16 ----------
__global__ void conv_bn_qkv(const float* __restrict__ x,
    const float* __restrict__ wq, const float* __restrict__ sq, const float* __restrict__ bq,
    const float* __restrict__ mq, const float* __restrict__ vq,
    const float* __restrict__ wk, const float* __restrict__ sk, const float* __restrict__ bk,
    const float* __restrict__ mk, const float* __restrict__ vk,
    const float* __restrict__ wv, const float* __restrict__ sv, const float* __restrict__ bv,
    const float* __restrict__ mv, const float* __restrict__ vv,
    bf16_t* __restrict__ qin, bf16_t* __restrict__ kin, bf16_t* __restrict__ vin) {
  const int idx = blockIdx.x * blockDim.x + threadIdx.x;   // grid exactly B*N*C
  const int c  = idx % CDIM;
  const int np = (idx / CDIM) & (NPOS - 1);
  const int b  = idx / (CDIM * NPOS);
  const int hh = np >> 5, ww = np & 31;
  float aq = 0.f, ak = 0.f, av = 0.f;
#pragma unroll
  for (int dh = 0; dh < 3; ++dh) {
    const int h2 = hh + dh - 1;
#pragma unroll
    for (int dw = 0; dw < 3; ++dw) {
      const int w2 = ww + dw - 1;
      float xv = 0.f;
      if (h2 >= 0 && h2 < 32 && w2 >= 0 && w2 < 32)
        xv = x[((size_t)b * NPOS + (h2 << 5) + w2) * CDIM + c];   // coalesced over c
      const int wi = c * 9 + dh * 3 + dw;
      aq = fmaf(wq[wi], xv, aq);
      ak = fmaf(wk[wi], xv, ak);
      av = fmaf(wv[wi], xv, av);
    }
  }
  const float iq = sq[c] * rsqrtf(vq[c] + 1e-5f);
  const float ik = sk[c] * rsqrtf(vk[c] + 1e-5f);
  const float iv = sv[c] * rsqrtf(vv[c] + 1e-5f);
  qin[idx] = (bf16_t)(aq * iq + (bq[c] - mq[c] * iq));
  kin[idx] = (bf16_t)(ak * ik + (bk[c] - mk[c] * ik));
  vin[idx] = (bf16_t)(av * iv + (bv[c] - mv[c] * iv));
}

// ---------- kernel 2: NT GEMM  Y[M x 384] = A[M x 384] @ W[384 x 384]^T ----------
// mode 0: store bf16 as q/k head layout [B, NH, N, 64]
// mode 1: store bf16 as v transposed   [B, NH, 64, N]
// mode 2: store f32 row-major [M x 384] + bias (final projection)
__global__ void gemm_nt(const bf16_t* __restrict__ A, const bf16_t* __restrict__ W,
                        void* __restrict__ Out, const float* __restrict__ bias, int mode) {
  const int K = CDIM, N = CDIM;
  const int wid = (blockIdx.x * blockDim.x + threadIdx.x) >> 5;
  const int nw = N / 32;                 // 12 column groups
  const int m0 = (wid / nw) * 32;
  const int n0 = (wid % nw) * 32;
  v8f acc[2][2] = {};
  for (int k0 = 0; k0 < K; k0 += 32) {
    v16bf a0 = load_afrag(A, K, m0, k0);
    v16bf a1 = load_afrag(A, K, m0 + 16, k0);
    v16bf b0 = load_bfrag(W, K, n0, k0);
    v16bf b1 = load_bfrag(W, K, n0 + 16, k0);
    acc[0][0] = wmma_bf16(a0, b0, acc[0][0]);
    acc[0][1] = wmma_bf16(a0, b1, acc[0][1]);
    acc[1][0] = wmma_bf16(a1, b0, acc[1][0]);
    acc[1][1] = wmma_bf16(a1, b1, acc[1][1]);
  }
  const int lane = threadIdx.x & 31;
  const int g = lane >> 4, r = lane & 15;
#pragma unroll
  for (int ti = 0; ti < 2; ++ti)
#pragma unroll
    for (int tj = 0; tj < 2; ++tj)
#pragma unroll
      for (int j = 0; j < 8; ++j) {
        const int m = m0 + 16 * ti + j + 8 * g;       // C/D layout: VGPR j -> row, lane -> col
        const int n = n0 + 16 * tj + r;
        const float val = acc[ti][tj][j];
        if (mode == 2) {
          ((float*)Out)[(size_t)m * N + n] = val + bias[n];
        } else {
          const int b = m >> 10, np = m & (NPOS - 1);
          const int nh = n >> 6, d = n & (HDIM - 1);
          size_t idx;
          if (mode == 0) idx = (((size_t)(b * NHEADS + nh)) * NPOS + np) * HDIM + d;
          else           idx = (((size_t)(b * NHEADS + nh)) * HDIM + d) * NPOS + np;
          ((bf16_t*)Out)[idx] = (bf16_t)val;
        }
      }
}

// ---------- kernel 3: flash attention, one wave per (b, head, 16-query block) ----------
// Computes scores^T = K_blk(16x64) @ Q^T(64x16) so softmax over keys is per-lane.
// Second GEMM: out^T(64x16) += V^T(16x32) @ P^T(32x16).
__global__ void attn_flash(const bf16_t* __restrict__ q, const bf16_t* __restrict__ k,
                           const bf16_t* __restrict__ vT, bf16_t* __restrict__ aout) {
  const int gw = (blockIdx.x * blockDim.x + threadIdx.x) >> 5;  // exactly B*NH*64 waves
  const int qb = gw & 63;
  const int h  = (gw >> 6) % NHEADS;
  const int b  = gw / (64 * NHEADS);
  const int q0 = qb << 4;
  const bf16_t* qp = q  + ((size_t)(b * NHEADS + h)) * NPOS * HDIM;
  const bf16_t* kp = k  + ((size_t)(b * NHEADS + h)) * NPOS * HDIM;
  const bf16_t* vp = vT + ((size_t)(b * NHEADS + h)) * HDIM * NPOS;

  // Q^T as B-fragments: lane = query column, per-lane d-contiguous (= q row).
  const v16bf qf0 = load_bfrag(qp, HDIM, q0, 0);
  const v16bf qf1 = load_bfrag(qp, HDIM, q0, 32);

  v8f acc0 = {}, acc1 = {}, acc2 = {}, acc3 = {};   // out^T tiles, d = 0..63
  float Mr = -1e30f, Lr = 0.f;                      // per-lane running max / sum (q-indexed)
  const int g = (threadIdx.x >> 4) & 1;

  for (int kb = 0; kb < NPOS; kb += 32) {
    v8f s0 = {}, s1 = {};
    s0 = wmma_bf16(load_afrag(kp, HDIM, kb,      0), qf0, s0);
    s0 = wmma_bf16(load_afrag(kp, HDIM, kb,     32), qf1, s0);
    s1 = wmma_bf16(load_afrag(kp, HDIM, kb + 16, 0), qf0, s1);
    s1 = wmma_bf16(load_afrag(kp, HDIM, kb + 16,32), qf1, s1);

    float mb = -1e30f;
#pragma unroll
    for (int j = 0; j < 8; ++j) {
      s0[j] *= 0.125f; s1[j] *= 0.125f;             // 1/sqrt(64)
      mb = fmaxf(mb, fmaxf(s0[j], s1[j]));
    }
    mb = fmaxf(mb, __shfl_xor(mb, 16, 32));         // merge the two lane halves
    const float Mn = fmaxf(Mr, mb);
    const float alpha = __expf(Mr - Mn);
    Mr = Mn;

    float p0[8], p1[8], ps = 0.f;
#pragma unroll
    for (int j = 0; j < 8; ++j) {
      p0[j] = __expf(s0[j] - Mn);
      p1[j] = __expf(s1[j] - Mn);
      ps += p0[j] + p1[j];
    }
    Lr = Lr * alpha + ps;                           // per-lane partial over its key half

    // Build P^T B-fragment (K=32 keys): lane group g needs all 16 rows of tile g,
    // half of which live in the xor-16 partner lane.
    v16bf pf;
#pragma unroll
    for (int j = 0; j < 8; ++j) {
      const float o0 = __shfl_xor(p0[j], 16, 32);
      const float o1 = __shfl_xor(p1[j], 16, 32);
      pf[j]     = (bf16_t)(g ? o1    : p0[j]);      // K local = j     (tile g rows 0..7)
      pf[j + 8] = (bf16_t)(g ? p1[j] : o0);         // K local = j + 8 (tile g rows 8..15)
    }

#pragma unroll
    for (int j = 0; j < 8; ++j) {
      acc0[j] *= alpha; acc1[j] *= alpha; acc2[j] *= alpha; acc3[j] *= alpha;
    }
    acc0 = wmma_bf16(load_afrag(vp, NPOS,  0, kb), pf, acc0);
    acc1 = wmma_bf16(load_afrag(vp, NPOS, 16, kb), pf, acc1);
    acc2 = wmma_bf16(load_afrag(vp, NPOS, 32, kb), pf, acc2);
    acc3 = wmma_bf16(load_afrag(vp, NPOS, 48, kb), pf, acc3);
  }

  Lr += __shfl_xor(Lr, 16, 32);                     // combine the two key-halves
  const float inv = 1.f / Lr;
  const int r = threadIdx.x & 15;
  bf16_t* op = aout + ((size_t)b * NPOS + q0 + r) * CDIM + h * HDIM;
#pragma unroll
  for (int j = 0; j < 8; ++j) {
    op[ 0 + j + 8 * g] = (bf16_t)(acc0[j] * inv);
    op[16 + j + 8 * g] = (bf16_t)(acc1[j] * inv);
    op[32 + j + 8 * g] = (bf16_t)(acc2[j] * inv);
    op[48 + j + 8 * g] = (bf16_t)(acc3[j] * inv);
  }
}

// ---------------------------------------------------------------------------
extern "C" void kernel_launch(void* const* d_in, const int* in_sizes, int n_in,
                              void* d_out, int out_size, void* d_ws, size_t ws_size,
                              hipStream_t stream) {
  (void)in_sizes; (void)n_in; (void)out_size; (void)ws_size;
  const float* x  = (const float*)d_in[0];
  const float* cw[3] = {(const float*)d_in[1], (const float*)d_in[6],  (const float*)d_in[11]};
  const float* bs[3] = {(const float*)d_in[2], (const float*)d_in[7],  (const float*)d_in[12]};
  const float* bb[3] = {(const float*)d_in[3], (const float*)d_in[8],  (const float*)d_in[13]};
  const float* bm[3] = {(const float*)d_in[4], (const float*)d_in[9],  (const float*)d_in[14]};
  const float* bv[3] = {(const float*)d_in[5], (const float*)d_in[10], (const float*)d_in[15]};
  const float* Wq = (const float*)d_in[16];
  const float* Wk = (const float*)d_in[17];
  const float* Wv = (const float*)d_in[18];
  const float* Wp = (const float*)d_in[19];
  const float* bp = (const float*)d_in[20];

  const size_t MROWS = (size_t)NB * NPOS;                         // 16384
  const size_t szW   = (size_t)CDIM * CDIM * sizeof(bf16_t);      // 288 KB
  const size_t szAct = MROWS * CDIM * sizeof(bf16_t);             // 12.6 MB

  char* ws = (char*)d_ws;
  size_t off = 0;
  auto take = [&](size_t bytes) -> void* {
    void* p = ws + off;
    off += (bytes + 255) & ~(size_t)255;
    return p;
  };
  bf16_t* wqb = (bf16_t*)take(szW);
  bf16_t* wkb = (bf16_t*)take(szW);
  bf16_t* wvb = (bf16_t*)take(szW);
  bf16_t* wpb = (bf16_t*)take(szW);
  bf16_t* qin = (bf16_t*)take(szAct);
  bf16_t* kin = (bf16_t*)take(szAct);
  bf16_t* vin = (bf16_t*)take(szAct);
  bf16_t* qh  = (bf16_t*)take(szAct);   // [B, NH, N, 64]
  bf16_t* kh  = (bf16_t*)take(szAct);   // [B, NH, N, 64]
  bf16_t* vTh = (bf16_t*)take(szAct);   // [B, NH, 64, N]
  bf16_t* aout = qin;                   // qin is dead after the q-GEMM; reuse for attention output

  cvt_weights<<<(CDIM * CDIM) / 256, 256, 0, stream>>>(Wq, Wk, Wv, Wp, wqb, wkb, wvb, wpb);

  conv_bn_qkv<<<(int)((MROWS * CDIM) / 256), 256, 0, stream>>>(
      x,
      cw[0], bs[0], bb[0], bm[0], bv[0],
      cw[1], bs[1], bb[1], bm[1], bv[1],
      cw[2], bs[2], bb[2], bm[2], bv[2],
      qin, kin, vin);

  // (M/32)*(N/32) = 512*12 = 6144 waves -> 768 blocks of 256 threads
  gemm_nt<<<768, 256, 0, stream>>>(qin, wqb, qh,  nullptr, 0);
  gemm_nt<<<768, 256, 0, stream>>>(kin, wkb, kh,  nullptr, 0);
  gemm_nt<<<768, 256, 0, stream>>>(vin, wvb, vTh, nullptr, 1);

  // B*NH*64 = 6144 waves -> 768 blocks
  attn_flash<<<768, 256, 0, stream>>>(qh, kh, vTh, aout);

  gemm_nt<<<768, 256, 0, stream>>>(aout, wpb, d_out, bp, 2);
}
